// GCN4MAML_23605140259074
// MI455X (gfx1250) — matrix operand order repaired and checked by reading, not compiled, not run on hardware
//
#include <hip/hip_runtime.h>
#include <hip/hip_bf16.h>
#include <math.h>

// ---------------- problem constants (match reference) ----------------
#define G_GRAPHS 128
#define N0 512
#define DEG 8
#define F_IN 64
#define H 128
#define C_CLS 30
#define E_TOT (G_GRAPHS * N0 * DEG)   // 524288
#define K1 (N0 / 2)                   // 256
#define K2 (N0 / 4)                   // 128
#define K3 (N0 / 8)                   // 64
#define NEG_SLOPE 0.1f

typedef __attribute__((ext_vector_type(16))) _Float16 v16h;
typedef __attribute__((ext_vector_type(8)))  _Float16 v8h;
typedef __attribute__((ext_vector_type(8)))  float    v8f;

__device__ __forceinline__ float lrelu(float v) {
    return v > 0.0f ? v : NEG_SLOPE * v;
}

// ---------------- utility fills / copies / conversions ----------------
__global__ void fill_f32(float* p, float v, int n) {
    int i = blockIdx.x * blockDim.x + threadIdx.x;
    if (i < n) p[i] = v;
}
__global__ void fill_i32(int* p, int v, int n) {
    int i = blockIdx.x * blockDim.x + threadIdx.x;
    if (i < n) p[i] = v;
}
__global__ void copy_f32(const float* __restrict__ src, float* __restrict__ dst, int n) {
    int i = blockIdx.x * blockDim.x + threadIdx.x;
    if (i < n) dst[i] = src[i];
}
__global__ void cvt_f32_f16(const float* __restrict__ src, _Float16* __restrict__ dst, int n) {
    int i = blockIdx.x * blockDim.x + threadIdx.x;
    if (i < n) dst[i] = (_Float16)src[i];
}
// Wt16[n, k] = (f16) W[k, n]   (W row-major KxN -> Wt col-major-of-W = [N,K])
__global__ void cvt_weight_t(const float* __restrict__ W, _Float16* __restrict__ Wt,
                             int K, int N) {
    int idx = blockIdx.x * blockDim.x + threadIdx.x;
    if (idx >= N * K) return;
    int n = idx / K, k = idx % K;
    Wt[idx] = (_Float16)W[k * N + n];
}

// ---------------- WMMA GEMM: C[M,N] = A16[M,K] @ W[K,N] (+bias, +lrelu) ----------------
// A16: f16 row-major [M,K]. Wt16: f16 [N,K] (transposed weights, contiguous K per column).
// Block: 256 threads = 8 waves; each wave owns one 16-row M-tile of a 128-row super-tile;
// all 8 waves share one 16-col N-tile whose weights are staged once in LDS.
// Requires M%128==0 or M==128k, K%32==0, N%16==0. EXEC all-ones everywhere.
__global__ void gemm_wmma_f16(const _Float16* __restrict__ A16,
                              const _Float16* __restrict__ Wt16,
                              const float* __restrict__ bias, float* __restrict__ C,
                              int M, int K, int N, int act) {
    extern __shared__ _Float16 ldsW[];     // [16][K+8] padded to spread banks
    const int S = K + 8;                   // padded LDS row stride (halfs)

    const int tid  = threadIdx.x;          // 0..255
    const int wave = tid >> 5;             // 0..7
    const int lane = tid & 31;
    const int n0 = blockIdx.x * 16;
    const int m0 = blockIdx.y * 128 + wave * 16;

    // ---- stage the 16-column weight tile (n0..n0+15, all K) into LDS ----
    for (int i = tid; i < 16 * K; i += 256) {
        int col = i / K, k = i - col * K;
        ldsW[col * S + k] = Wt16[(size_t)(n0 + col) * K + k];
    }
    __syncthreads();

    const int lh = lane & 15;              // 0..15
    const int hi = lane >> 4;              // 0 or 1
    const _Float16* arow_p = A16 + (size_t)(m0 + lh) * K;
    const _Float16* bcol_p = ldsW + lh * S;

    v8f acc = {};
    for (int kk = 0; kk < K; kk += 32) {
        // A fragment: lane<16 -> K = kk+0..7 & kk+16..23 ; lane>=16 -> kk+8..15 & kk+24..31
        const int kA = kk + (hi ? 8 : 0);
        v8h alo = *(const v8h*)(arow_p + kA);
        v8h ahi = *(const v8h*)(arow_p + kA + 16);
        __builtin_prefetch(arow_p + kA + 32, 0, 0);
        v16h a;
#pragma unroll
        for (int j = 0; j < 8; ++j) { a[j] = alo[j]; a[8 + j] = ahi[j]; }

        // B fragment from LDS: lane<16 -> K = kk+0..7 & kk+16..23 at col lh ; lane>=16 -> kk+8..15 & kk+24..31
        const int kB = kk + (hi ? 8 : 0);
        v8h blo = *(const v8h*)(bcol_p + kB);
        v8h bhi = *(const v8h*)(bcol_p + kB + 16);
        v16h b;
#pragma unroll
        for (int j = 0; j < 8; ++j) { b[j] = blo[j]; b[8 + j] = bhi[j]; }

        acc = __builtin_amdgcn_wmma_f32_16x16x32_f16(
            /*neg_a=*/false, a, /*neg_b=*/false, b,
            /*c_mod=*/(short)0, acc, /*reuse_a=*/false, /*reuse_b=*/false);
    }

    // C tile store: lane<16 -> rows m0+0..7, lane>=16 -> rows m0+8..15; col = n0+lh
    const int rbase = m0 + (hi ? 8 : 0);
    const int bcol = n0 + lh;
    const float bv = bias ? bias[bcol] : 0.0f;
#pragma unroll
    for (int r = 0; r < 8; ++r) {
        float v = acc[r] + bv;
        if (act) v = lrelu(v);
        C[(size_t)(rbase + r) * N + bcol] = v;
    }
}

// ---------------- degree accumulation: deg[idx[e]] += w(e) ----------------
__global__ void deg_kernel(const int* __restrict__ idx, const float* __restrict__ ew,
                           float* __restrict__ deg, int E) {
    int e = blockIdx.x * blockDim.x + threadIdx.x;
    if (e >= E) return;
    float w = ew ? ew[e] : 1.0f;
    if (w != 0.0f) atomicAdd(&deg[idx[e]], w);
}

// dinv = (deg+addv) > 0 ? rsqrt(deg+addv) : 0
__global__ void dinv_kernel(const float* __restrict__ deg, float* __restrict__ dinv,
                            int n, float addv) {
    int i = blockIdx.x * blockDim.x + threadIdx.x;
    if (i >= n) return;
    float d = deg[i] + addv;
    dinv[i] = d > 0.0f ? rsqrtf(d) : 0.0f;
}

// out[col[e]] += sign * dinv[row]*w*dinv[col] * h[row]   (4 feats per thread, H=128)
__global__ void scatter_kernel(const int* __restrict__ row, const int* __restrict__ col,
                               const float* __restrict__ ew, const float* __restrict__ dinv,
                               const float* __restrict__ h, float* __restrict__ out,
                               int E, float sign) {
    int idx = blockIdx.x * blockDim.x + threadIdx.x;
    int e = idx >> 5;           // H/4 = 32 chunks per edge
    int ch = idx & 31;
    if (e >= E) return;
    float w = ew ? ew[e] : 1.0f;
    if (w == 0.0f) return;
    int r = row[e], c = col[e];
    float nrm = sign * dinv[r] * w * dinv[c];
    const float4 hv = *(const float4*)(h + (size_t)r * H + ch * 4);
    float* op = out + (size_t)c * H + ch * 4;
    atomicAdd(op + 0, nrm * hv.x);
    atomicAdd(op + 1, nrm * hv.y);
    atomicAdd(op + 2, nrm * hv.z);
    atomicAdd(op + 3, nrm * hv.w);
}

// out[n,f] = lrelu(out[n,f] + dinv[n]^2 * hlin[n,f] + b[f])
__global__ void selfloop_bias_act(const float* __restrict__ hlin, const float* __restrict__ dinv,
                                  const float* __restrict__ b, float* __restrict__ out, int nodes) {
    int idx = blockIdx.x * blockDim.x + threadIdx.x;
    if (idx >= nodes * H) return;
    int n = idx >> 7, f = idx & 127;
    float di = dinv[n];
    out[idx] = lrelu(out[idx] + di * di * hlin[idx] + b[f]);
}

// invnorm = 1/||pw||  (one block of 128 threads)
__global__ void pwnorm_kernel(const float* __restrict__ pw, float* __restrict__ invnorm) {
    __shared__ float s[H];
    int t = threadIdx.x;
    float v = pw[t];
    s[t] = v * v;
    __syncthreads();
    for (int o = 64; o > 0; o >>= 1) {
        if (t < o) s[t] += s[t + o];
        __syncthreads();
    }
    if (t == 0) invnorm[0] = rsqrtf(s[0]);
}

// score[n] = dot(h[n], pw) * invnorm
__global__ void score_kernel(const float* __restrict__ h, const float* __restrict__ pw,
                             const float* __restrict__ invnorm, float* __restrict__ score,
                             int nodes) {
    int n = blockIdx.x * blockDim.x + threadIdx.x;
    if (n >= nodes) return;
    const float* hp = h + (size_t)n * H;
    float s = 0.0f;
#pragma unroll 4
    for (int f = 0; f < H; ++f) s += hp[f] * pw[f];
    score[n] = s * invnorm[0];
}

// per-graph top-k via in-LDS bitonic sort (descending score, tie -> smaller idx).
// gridDim.x = G, blockDim.x = Ncur (power of two, <=512).
__global__ void topk_kernel(const float* __restrict__ score, float* __restrict__ gate,
                            int* __restrict__ perm, int* __restrict__ old2new,
                            int Ncur, int k) {
    __shared__ float s[512];
    __shared__ int   id[512];
    int g = blockIdx.x, tid = threadIdx.x;
    s[tid] = score[g * Ncur + tid];
    id[tid] = tid;
    __syncthreads();
    for (int ks = 2; ks <= Ncur; ks <<= 1) {
        for (int j = ks >> 1; j > 0; j >>= 1) {
            int ixj = tid ^ j;
            if (ixj > tid) {
                float s0 = s[tid], s1 = s[ixj];
                int i0 = id[tid], i1 = id[ixj];
                bool worse = (s0 < s1) || (s0 == s1 && i0 > i1);
                bool desc = ((tid & ks) == 0);
                if (desc ? worse : !worse) {
                    s[tid] = s1; s[ixj] = s0;
                    id[tid] = i1; id[ixj] = i0;
                }
            }
            __syncthreads();
        }
    }
    if (tid < k) {
        int old = g * Ncur + id[tid];
        perm[g * k + tid] = old;
        gate[g * k + tid] = tanhf(s[tid]);
        old2new[old] = g * k + tid;
    }
}

// xnew[i,f] = h[perm[i],f] * gate[i]; also emit f16 copy for the next WMMA GEMM
__global__ void gather_gate(const float* __restrict__ h, const int* __restrict__ perm,
                            const float* __restrict__ gate, float* __restrict__ xnew,
                            _Float16* __restrict__ xnew16, int newnodes) {
    int idx = blockIdx.x * blockDim.x + threadIdx.x;
    if (idx >= newnodes * H) return;
    int i = idx >> 7, f = idx & 127;
    float v = h[(size_t)perm[i] * H + f] * gate[i];
    xnew[idx] = v;
    xnew16[idx] = (_Float16)v;
}

// remap edges through old2new; invalid edges -> (0,0,w=0)
__global__ void remap_edges(const int* __restrict__ row, const int* __restrict__ col,
                            const float* __restrict__ ew, const int* __restrict__ old2new,
                            int* __restrict__ nrow, int* __restrict__ ncol,
                            float* __restrict__ new_ew, int E) {
    int e = blockIdx.x * blockDim.x + threadIdx.x;
    if (e >= E) return;
    float w = ew ? ew[e] : 1.0f;
    int nr = old2new[row[e]];
    int nc = old2new[col[e]];
    bool valid = (nr >= 0) && (nc >= 0) && (w > 0.5f);
    nrow[e] = valid ? nr : 0;
    ncol[e] = valid ? nc : 0;
    new_ew[e] = valid ? 1.0f : 0.0f;
}

// readout: out[g, f] = max_j h[g*k+j, f]; out[g, H+f] = mean_j. grid=G, block=H.
__global__ void readout_kernel(const float* __restrict__ h, float* __restrict__ out, int k) {
    int g = blockIdx.x, f = threadIdx.x;
    const float* hp = h + (size_t)g * k * H + f;
    float mx = -3.4e38f, sm = 0.0f;
    for (int j = 0; j < k; ++j) {
        float v = hp[(size_t)j * H];
        mx = fmaxf(mx, v);
        sm += v;
    }
    out[(size_t)g * (2 * H) + f] = mx;
    out[(size_t)g * (2 * H) + H + f] = sm / (float)k;
}

// g = lrelu(x1) + lrelu(x2) + lrelu(x3)
__global__ void combine_readouts(const float* __restrict__ x1, const float* __restrict__ x2,
                                 const float* __restrict__ x3, float* __restrict__ g, int n) {
    int i = blockIdx.x * blockDim.x + threadIdx.x;
    if (i < n) g[i] = lrelu(x1[i]) + lrelu(x2[i]) + lrelu(x3[i]);
}

// per-node sum of |inf| accumulated into a scalar
__global__ void abs_sum_kernel(const float* __restrict__ inf, float* __restrict__ accum,
                               int nodes) {
    int n = blockIdx.x * blockDim.x + threadIdx.x;
    if (n >= nodes) return;
    const float* p = inf + (size_t)n * H;
    float s = 0.0f;
#pragma unroll 4
    for (int f = 0; f < H; ++f) s += fabsf(p[f]);
    atomicAdd(accum, s);
}

// logits: out[g,c] = sum_j gm2[g,j]*lw3[j,c] + lb3[c]  (K=64, N=30)
__global__ void logits_kernel(const float* __restrict__ gm2, const float* __restrict__ lw3,
                              const float* __restrict__ lb3, float* __restrict__ out) {
    int idx = blockIdx.x * blockDim.x + threadIdx.x;
    if (idx >= G_GRAPHS * C_CLS) return;
    int g = idx / C_CLS, c = idx % C_CLS;
    const float* gp = gm2 + g * (H / 2);
    float s = lb3[c];
#pragma unroll 4
    for (int j = 0; j < H / 2; ++j) s += gp[j] * lw3[j * C_CLS + c];
    out[idx] = s;
}

__global__ void finalize_kernel(float* __restrict__ out_scalar, const float* __restrict__ accum,
                                float invN) {
    if (threadIdx.x == 0 && blockIdx.x == 0) out_scalar[0] = accum[0] * invN;
}

// ---------------- host orchestration ----------------
static inline int nb(long n) { return (int)((n + 255) / 256); }

extern "C" void kernel_launch(void* const* d_in, const int* in_sizes, int n_in,
                              void* d_out, int out_size, void* d_ws, size_t ws_size,
                              hipStream_t stream) {
    (void)in_sizes; (void)n_in; (void)out_size; (void)ws_size;

    const float* x   = (const float*)d_in[0];
    const int*   row = (const int*)d_in[1];
    const int*   col = (const int*)d_in[2];
    const float* W1  = (const float*)d_in[3];
    const float* b1  = (const float*)d_in[4];
    const float* W2  = (const float*)d_in[5];
    const float* b2  = (const float*)d_in[6];
    const float* W3  = (const float*)d_in[7];
    const float* b3  = (const float*)d_in[8];
    const float* pw1 = (const float*)d_in[9];
    const float* pw2 = (const float*)d_in[10];
    const float* pw3 = (const float*)d_in[11];
    const float* lw1 = (const float*)d_in[12];
    const float* lb1 = (const float*)d_in[13];
    const float* lw2 = (const float*)d_in[14];
    const float* lb2 = (const float*)d_in[15];
    const float* lw3 = (const float*)d_in[16];
    const float* lb3 = (const float*)d_in[17];
    float* out = (float*)d_out;

    // ---- workspace carve-out ----
    char* base = (char*)d_ws;
    size_t off = 0;
    auto alloc = [&](size_t bytes) -> char* {
        char* p = base + off;
        off = (off + bytes + 255) & ~(size_t)255;
        return p;
    };
    const int NODES1 = G_GRAPHS * N0;   // 65536
    const int NODES2 = G_GRAPHS * K1;   // 32768
    const int NODES3 = G_GRAPHS * K2;   // 16384
    const int NODES4 = G_GRAPHS * K3;   // 8192

    float*     bufA   = (float*)alloc((size_t)NODES1 * H * 4);   // hlin / inf
    float*     bufB   = (float*)alloc((size_t)NODES1 * H * 4);   // h aggregated
    float*     bufP   = (float*)alloc((size_t)NODES2 * H * 4);   // pooled features (f32)
    _Float16*  a16    = (_Float16*)alloc((size_t)NODES1 * F_IN * 2); // f16 GEMM A (max 65536*64)
    _Float16*  w16    = (_Float16*)alloc((size_t)(2 * H) * H * 2);   // f16 transposed weights
    int*   rowA   = (int*)alloc((size_t)E_TOT * 4);
    int*   colA   = (int*)alloc((size_t)E_TOT * 4);
    float* ewA    = (float*)alloc((size_t)E_TOT * 4);
    int*   rowB   = (int*)alloc((size_t)E_TOT * 4);
    int*   colB   = (int*)alloc((size_t)E_TOT * 4);
    float* ewB    = (float*)alloc((size_t)E_TOT * 4);
    float* deg    = (float*)alloc((size_t)NODES1 * 4);
    float* dinv   = (float*)alloc((size_t)NODES1 * 4);
    float* score  = (float*)alloc((size_t)NODES1 * 4);
    float* gate   = (float*)alloc((size_t)NODES2 * 4);
    int*   perm   = (int*)alloc((size_t)NODES2 * 4);
    int*   o2n    = (int*)alloc((size_t)NODES1 * 4);
    float* x1r    = (float*)alloc((size_t)G_GRAPHS * 2 * H * 4);
    float* x2r    = (float*)alloc((size_t)G_GRAPHS * 2 * H * 4);
    float* x3r    = (float*)alloc((size_t)G_GRAPHS * 2 * H * 4);
    float* gcomb  = (float*)alloc((size_t)G_GRAPHS * 2 * H * 4);
    float* gm1    = (float*)alloc((size_t)G_GRAPHS * H * 4);
    float* gm2    = (float*)alloc((size_t)G_GRAPHS * (H / 2) * 4);
    float* invnrm = (float*)alloc(4);
    float* accum  = (float*)alloc(4);

    auto sh = [](int K) { return (size_t)16 * (K + 8) * sizeof(_Float16); };

    // ================= stage 1 =================
    // hlin = x @ W1  (M=65536, K=64, N=128)
    cvt_f32_f16<<<nb((long)NODES1 * F_IN), 256, 0, stream>>>(x, a16, NODES1 * F_IN);
    cvt_weight_t<<<nb((long)F_IN * H), 256, 0, stream>>>(W1, w16, F_IN, H);
    gemm_wmma_f16<<<dim3(H / 16, NODES1 / 128), 256, sh(F_IN), stream>>>(
        a16, w16, nullptr, bufA, NODES1, F_IN, H, 0);
    fill_f32<<<nb(NODES1), 256, 0, stream>>>(deg, 0.0f, NODES1);
    fill_f32<<<nb((long)NODES1 * H), 256, 0, stream>>>(bufB, 0.0f, NODES1 * H);
    deg_kernel<<<nb(E_TOT), 256, 0, stream>>>(col, nullptr, deg, E_TOT);
    dinv_kernel<<<nb(NODES1), 256, 0, stream>>>(deg, dinv, NODES1, 1.0f);
    scatter_kernel<<<nb((long)E_TOT * 32), 256, 0, stream>>>(row, col, nullptr, dinv,
                                                             bufA, bufB, E_TOT, 1.0f);
    selfloop_bias_act<<<nb((long)NODES1 * H), 256, 0, stream>>>(bufA, dinv, b1, bufB, NODES1);
    // pool 1
    pwnorm_kernel<<<1, H, 0, stream>>>(pw1, invnrm);
    score_kernel<<<nb(NODES1), 256, 0, stream>>>(bufB, pw1, invnrm, score, NODES1);
    fill_i32<<<nb(NODES1), 256, 0, stream>>>(o2n, -1, NODES1);
    topk_kernel<<<G_GRAPHS, N0, 0, stream>>>(score, gate, perm, o2n, N0, K1);
    gather_gate<<<nb((long)NODES2 * H), 256, 0, stream>>>(bufB, perm, gate, bufP, a16, NODES2);
    remap_edges<<<nb(E_TOT), 256, 0, stream>>>(row, col, nullptr, o2n, rowA, colA, ewA, E_TOT);
    readout_kernel<<<G_GRAPHS, H, 0, stream>>>(bufP, x1r, K1);

    // ================= stage 2 =================
    cvt_weight_t<<<nb((long)H * H), 256, 0, stream>>>(W2, w16, H, H);
    gemm_wmma_f16<<<dim3(H / 16, NODES2 / 128), 256, sh(H), stream>>>(
        a16, w16, nullptr, bufA, NODES2, H, H, 0);
    fill_f32<<<nb(NODES2), 256, 0, stream>>>(deg, 0.0f, NODES2);
    fill_f32<<<nb((long)NODES2 * H), 256, 0, stream>>>(bufB, 0.0f, NODES2 * H);
    deg_kernel<<<nb(E_TOT), 256, 0, stream>>>(colA, ewA, deg, E_TOT);
    dinv_kernel<<<nb(NODES2), 256, 0, stream>>>(deg, dinv, NODES2, 1.0f);
    scatter_kernel<<<nb((long)E_TOT * 32), 256, 0, stream>>>(rowA, colA, ewA, dinv,
                                                             bufA, bufB, E_TOT, 1.0f);
    selfloop_bias_act<<<nb((long)NODES2 * H), 256, 0, stream>>>(bufA, dinv, b2, bufB, NODES2);
    // pool 2
    pwnorm_kernel<<<1, H, 0, stream>>>(pw2, invnrm);
    score_kernel<<<nb(NODES2), 256, 0, stream>>>(bufB, pw2, invnrm, score, NODES2);
    fill_i32<<<nb(NODES2), 256, 0, stream>>>(o2n, -1, NODES2);
    topk_kernel<<<G_GRAPHS, K1, 0, stream>>>(score, gate, perm, o2n, K1, K2);
    gather_gate<<<nb((long)NODES3 * H), 256, 0, stream>>>(bufB, perm, gate, bufP, a16, NODES3);
    remap_edges<<<nb(E_TOT), 256, 0, stream>>>(rowA, colA, ewA, o2n, rowB, colB, ewB, E_TOT);
    readout_kernel<<<G_GRAPHS, H, 0, stream>>>(bufP, x2r, K2);

    // ================= stage 3 =================
    cvt_weight_t<<<nb((long)H * H), 256, 0, stream>>>(W3, w16, H, H);
    gemm_wmma_f16<<<dim3(H / 16, NODES3 / 128), 256, sh(H), stream>>>(
        a16, w16, nullptr, bufA, NODES3, H, H, 0);
    fill_f32<<<nb(NODES3), 256, 0, stream>>>(deg, 0.0f, NODES3);
    fill_f32<<<nb((long)NODES3 * H), 256, 0, stream>>>(bufB, 0.0f, NODES3 * H);
    deg_kernel<<<nb(E_TOT), 256, 0, stream>>>(colB, ewB, deg, E_TOT);
    dinv_kernel<<<nb(NODES3), 256, 0, stream>>>(deg, dinv, NODES3, 1.0f);
    scatter_kernel<<<nb((long)E_TOT * 32), 256, 0, stream>>>(rowB, colB, ewB, dinv,
                                                             bufA, bufB, E_TOT, 1.0f);
    selfloop_bias_act<<<nb((long)NODES3 * H), 256, 0, stream>>>(bufA, dinv, b3, bufB, NODES3);
    // pool 3
    pwnorm_kernel<<<1, H, 0, stream>>>(pw3, invnrm);
    score_kernel<<<nb(NODES3), 256, 0, stream>>>(bufB, pw3, invnrm, score, NODES3);
    fill_i32<<<nb(NODES3), 256, 0, stream>>>(o2n, -1, NODES3);
    topk_kernel<<<G_GRAPHS, K2, 0, stream>>>(score, gate, perm, o2n, K2, K3);
    gather_gate<<<nb((long)NODES4 * H), 256, 0, stream>>>(bufB, perm, gate, bufP, a16, NODES4);
    remap_edges<<<nb(E_TOT), 256, 0, stream>>>(rowB, colB, ewB, o2n, rowA, colA, ewA, E_TOT);
    readout_kernel<<<G_GRAPHS, H, 0, stream>>>(bufP, x3r, K3);

    // ---- info score on final pooled graph (deg accumulated at ROW here) ----
    fill_f32<<<nb(NODES4), 256, 0, stream>>>(deg, 0.0f, NODES4);
    deg_kernel<<<nb(E_TOT), 256, 0, stream>>>(rowA, ewA, deg, E_TOT);
    dinv_kernel<<<nb(NODES4), 256, 0, stream>>>(deg, dinv, NODES4, 0.0f);
    copy_f32<<<nb((long)NODES4 * H), 256, 0, stream>>>(bufP, bufA, NODES4 * H);
    scatter_kernel<<<nb((long)E_TOT * 32), 256, 0, stream>>>(rowA, colA, ewA, dinv,
                                                             bufP, bufA, E_TOT, -1.0f);
    fill_f32<<<1, 256, 0, stream>>>(accum, 0.0f, 1);
    abs_sum_kernel<<<nb(NODES4), 256, 0, stream>>>(bufA, accum, NODES4);

    // ================= classifier head =================
    combine_readouts<<<nb((long)G_GRAPHS * 2 * H), 256, 0, stream>>>(x1r, x2r, x3r, gcomb,
                                                                     G_GRAPHS * 2 * H);
    // g @ lw1 + lb1, lrelu  (M=128, K=256, N=128)
    cvt_f32_f16<<<nb((long)G_GRAPHS * 2 * H), 256, 0, stream>>>(gcomb, a16, G_GRAPHS * 2 * H);
    cvt_weight_t<<<nb((long)2 * H * H), 256, 0, stream>>>(lw1, w16, 2 * H, H);
    gemm_wmma_f16<<<dim3(H / 16, 1), 256, sh(2 * H), stream>>>(
        a16, w16, lb1, gm1, G_GRAPHS, 2 * H, H, 1);
    // gm1 @ lw2 + lb2, lrelu (M=128, K=128, N=64)
    cvt_f32_f16<<<nb((long)G_GRAPHS * H), 256, 0, stream>>>(gm1, a16, G_GRAPHS * H);
    cvt_weight_t<<<nb((long)H * (H / 2)), 256, 0, stream>>>(lw2, w16, H, H / 2);
    gemm_wmma_f16<<<dim3((H / 2) / 16, 1), 256, sh(H), stream>>>(
        a16, w16, lb2, gm2, G_GRAPHS, H, H / 2, 1);
    // logits (N=30 not a multiple of 16 -> scalar kernel)
    logits_kernel<<<nb((long)G_GRAPHS * C_CLS), 256, 0, stream>>>(gm2, lw3, lb3, out);
    finalize_kernel<<<1, 32, 0, stream>>>(out + G_GRAPHS * C_CLS, accum,
                                          1.0f / (float)NODES4);
}
